// MixtureOfDepthTransformer_48576080118320
// MI455X (gfx1250) — compile-verified, matrix operand order checked by reading, not run on hardware
//
#include <hip/hip_runtime.h>
#include <cstdint>
#include <cstddef>

// ---------------------------------------------------------------- constants
#define Dm    1024
#define Hh    16
#define HDh   64
#define DFFm  4096
#define NLm   6
#define Tm    2048
#define Bm    4
#define KSEL  1024   // CAP=0.5 -> top-1024 tokens per sequence
#define KBLK  32     // GEMM k-tile staged by TDM per stage

typedef __bf16 bf16_t;
typedef __attribute__((ext_vector_type(16))) __bf16 v16bf;
typedef __attribute__((ext_vector_type(8)))  float  v8f;
typedef __attribute__((ext_vector_type(4)))  unsigned int u32x4;
typedef __attribute__((ext_vector_type(8)))  int i32x8;
typedef __attribute__((ext_vector_type(4)))  int i32x4;

struct U4 { unsigned int x0, x1, x2, x3; };
union FragCast { U4 u[2]; v16bf v; };

// fp32 -> bf16, round to nearest even
__device__ __forceinline__ bf16_t f2bf(float f) {
  union { float f; unsigned u; } a; a.f = f;
  unsigned r = a.u + 0x7FFFu + ((a.u >> 16) & 1u);
  union { unsigned short s; bf16_t b; } o; o.s = (unsigned short)(r >> 16);
  return o.b;
}

// 16-bit A/B fragment (16x32, K-contiguous rows) per CDNA5 ISA layout:
// lane L: row = L%16 ; VGPR0-3 hold k = base..base+7, VGPR4-7 hold k = base+16..base+23,
// base = 0 (lanes 0-15) or 8 (lanes 16-31).  Two 16B chunks per lane.
__device__ __forceinline__ v16bf load_frag_k(const bf16_t* base, long stride) {
  int lane = threadIdx.x & 31;
  int r    = lane & 15;
  int kb   = (lane & 16) ? 8 : 0;
  const bf16_t* p = base + (long)r * stride + kb;
  FragCast f;
  f.u[0] = *reinterpret_cast<const U4*>(p);
  f.u[1] = *reinterpret_cast<const U4*>(p + 16);
  return f.v;
}

// B fragment where memory is [k][n] row-major, n contiguous (V tiles): gather loads.
__device__ __forceinline__ v16bf load_frag_n(const bf16_t* base, long stride) {
  int lane = threadIdx.x & 31;
  int n    = lane & 15;
  int kb   = (lane & 16) ? 8 : 0;
  v16bf f;
#pragma unroll
  for (int i = 0; i < 8; ++i) f[i] = base[(long)(kb + i) * stride + n];
#pragma unroll
  for (int i = 0; i < 8; ++i) f[8 + i] = base[(long)(kb + 16 + i) * stride + n];
  return f;
}

__device__ __forceinline__ v8f wmma_bf16(v16bf a, v16bf b, v8f c) {
  return __builtin_amdgcn_wmma_f32_16x16x32_bf16(false, a, false, b, (short)0, c,
                                                 false, false);
}

// ---------------------------------------------------------------- TDM 2D tile load
// DMA `rows` x `tileK` bf16 elements (row-major, row stride `strideElems`) from
// global memory into LDS at byte offset `ldsOff`.  Descriptor per CDNA5 ISA ch.8.
__device__ __forceinline__ void tdm_load_2d(unsigned ldsOff, const bf16_t* gptr,
                                            unsigned rows, unsigned tileK,
                                            unsigned tensorRows,
                                            unsigned long long strideElems) {
  unsigned long long ga = (unsigned long long)(size_t)gptr;
  u32x4 g0;
  g0[0] = 1u;                                       // count=1, user descriptor
  g0[1] = ldsOff;                                   // lds_addr (bytes)
  g0[2] = (unsigned)(ga & 0xFFFFFFFFull);           // global_addr[31:0]
  g0[3] = (unsigned)((ga >> 32) & 0x1FFFFFFull)     // global_addr[56:32]
        | (2u << 30);                               // type = 2 ("image")
  unsigned td0 = (unsigned)strideElems;             // tensor_dim0 (row length)
  unsigned long long st0 = strideElems;             // tensor_dim0_stride
  i32x8 g1;
  g1[0] = (int)(1u << 16);                          // wg_mask=0, data_size=1 (2B)
  g1[1] = (int)((td0 & 0xFFFFu) << 16);             // [63:48] = tensor_dim0[15:0]
  g1[2] = (int)((td0 >> 16) | ((tensorRows & 0xFFFFu) << 16));  // dim0 hi | dim1 lo
  g1[3] = (int)(((tensorRows >> 16) & 0xFFFFu) | (tileK << 16)); // dim1 hi | tile_dim0
  g1[4] = (int)(rows & 0xFFFFu);                    // tile_dim1 ; tile_dim2 = 0
  g1[5] = (int)(st0 & 0xFFFFFFFFull);               // stride0[31:0]
  g1[6] = (int)((st0 >> 32) & 0xFFFFull);           // stride0[47:32] ; stride1 lo = 0
  g1[7] = 0;                                        // stride1 hi = 0
  i32x4 z4 = {0, 0, 0, 0};
#if __clang_major__ >= 23
  i32x8 z8 = {0, 0, 0, 0, 0, 0, 0, 0};
  __builtin_amdgcn_tensor_load_to_lds(g0, g1, z4, z4, z8, 0);
#else
  __builtin_amdgcn_tensor_load_to_lds(g0, g1, z4, z4, 0);
#endif
}

// ---------------------------------------------------------------- fp32 -> bf16 copy
__global__ __launch_bounds__(256) void k_cvt(const float* __restrict__ s,
                                             bf16_t* __restrict__ d, long n) {
  long i = (long)blockIdx.x * blockDim.x + threadIdx.x;
  long st = (long)gridDim.x * blockDim.x;
  for (; i < n; i += st) d[i] = f2bf(s[i]);
}

// ---------------------------------------------------------------- bf16 WMMA GEMM
// out[M,N] = A[M,K] (bf16) * W[N,K]^T (bf16) + bias ; optional ReLU, f32/bf16 stores.
// block = 256 thr = 8 waves -> 64(M) x 256(N) tile; wave = 32x64 (8 WMMAs/stage).
// A/B tiles double-buffered in LDS, filled by the Tensor Data Mover (wave 0),
// consumed via ds_load fragments + v_wmma.
__global__ __launch_bounds__(256) void k_gemm(const bf16_t* __restrict__ A,
                                              const bf16_t* __restrict__ W,
                                              const float* __restrict__ bias,
                                              float* __restrict__ outF,
                                              bf16_t* __restrict__ outB,
                                              int M, int N, int K, int relu) {
  __shared__ __align__(16) bf16_t As[2][64 * KBLK];    //  2 x 4 KB
  __shared__ __align__(16) bf16_t Bs[2][256 * KBLK];   //  2 x 16 KB
  const int wave = threadIdx.x >> 5;
  const int lane = threadIdx.x & 31;
  const int wm = wave >> 2, wn = wave & 3;             // wm: 32-row half, wn: 64-col strip
  const int mblk = blockIdx.y * 64;
  const int nblk = blockIdx.x * 256;
  const int nk = K / KBLK;

  if (wave == 0) {   // stage 0 DMA
    tdm_load_2d((unsigned)(size_t)&As[0][0], A + (long)mblk * K, 64, KBLK,
                (unsigned)M, (unsigned long long)K);
    tdm_load_2d((unsigned)(size_t)&Bs[0][0], W + (long)nblk * K, 256, KBLK,
                (unsigned)N, (unsigned long long)K);
  }

  v8f z = {};
  v8f acc0[4] = {z, z, z, z};
  v8f acc1[4] = {z, z, z, z};

  for (int s = 0; s < nk; ++s) {
    if (wave == 0) {
      if (s + 1 < nk) {   // prefetch stage s+1 via TDM, then wait for stage s
        int kn = (s + 1) * KBLK;
        int bn = (s + 1) & 1;
        tdm_load_2d((unsigned)(size_t)&As[bn][0], A + (long)mblk * K + kn, 64, KBLK,
                    (unsigned)M, (unsigned long long)K);
        tdm_load_2d((unsigned)(size_t)&Bs[bn][0], W + (long)nblk * K + kn, 256, KBLK,
                    (unsigned)N, (unsigned long long)K);
        __builtin_amdgcn_s_wait_tensorcnt(2);
      } else {
        __builtin_amdgcn_s_wait_tensorcnt(0);
      }
    }
    __syncthreads();                       // stage s resident in LDS
    const bf16_t* at = &As[s & 1][(wm * 32) * KBLK];
    const bf16_t* bt = &Bs[s & 1][(wn * 64) * KBLK];
    v16bf a0 = load_frag_k(at, KBLK);
    v16bf a1 = load_frag_k(at + 16 * KBLK, KBLK);
#pragma unroll
    for (int j = 0; j < 4; ++j) {
      v16bf b = load_frag_k(bt + (long)j * 16 * KBLK, KBLK);
      acc0[j] = wmma_bf16(a0, b, acc0[j]);
      acc1[j] = wmma_bf16(a1, b, acc1[j]);
    }
    __syncthreads();                       // done reading buf before TDM reuses it
  }

  int col = lane & 15, rb = (lane & 16) ? 8 : 0;
#pragma unroll
  for (int j = 0; j < 4; ++j) {
    int n = nblk + wn * 64 + j * 16 + col;
    float bs = bias ? bias[n] : 0.0f;
#pragma unroll
    for (int r = 0; r < 8; ++r) {
      long m0 = mblk + wm * 32 + rb + r;
      float v0 = acc0[j][r] + bs;
      float v1 = acc1[j][r] + bs;
      if (relu) { v0 = fmaxf(v0, 0.0f); v1 = fmaxf(v1, 0.0f); }
      if (outF) { outF[m0 * N + n] = v0; outF[(m0 + 16) * N + n] = v1; }
      if (outB) { outB[m0 * N + n] = f2bf(v0); outB[(m0 + 16) * N + n] = f2bf(v1); }
    }
  }
}

// ---------------------------------------------------------------- flash attention
// qkv: [Btok, 3*Dm] bf16 (q|k|v), out: [Btok, Dm] bf16.  One wave per (b,h,qtile16).
__global__ __launch_bounds__(256) void k_attn(const bf16_t* __restrict__ qkv,
                                              bf16_t* __restrict__ out, int Tt) {
  __shared__ __align__(16) bf16_t pbuf[8][16 * 32];
  const long ld = 3 * Dm;
  int wave = threadIdx.x >> 5, lane = threadIdx.x & 31;
  int gw = blockIdx.x * 8 + wave;
  int numQT = Tt >> 4;
  int qt = gw % numQT;
  int h  = (gw / numQT) % Hh;
  int b  = gw / (numQT * Hh);
  const bf16_t* qp = qkv + (long)b * Tt * ld + (long)h * HDh;
  const bf16_t* kp = qp + Dm;
  const bf16_t* vp = qp + 2 * Dm;

  v16bf qa0 = load_frag_k(qp + (long)(qt * 16) * ld, ld);
  v16bf qa1 = load_frag_k(qp + (long)(qt * 16) * ld + 32, ld);

  v8f z = {};
  v8f oacc[4] = {z, z, z, z};
  float mrow[8], lrow[8];
#pragma unroll
  for (int r = 0; r < 8; ++r) { mrow[r] = -__builtin_inff(); lrow[r] = 0.0f; }

  const float scale = 0.125f;  // 1/sqrt(64)
  for (int kk = 0; kk < Tt; kk += 32) {
    if (kk + 32 < Tt) __builtin_prefetch(kp + (long)(kk + 32) * ld, 0, 3);
    // issue all global loads for this block up-front; latency hides under softmax
    v16bf kb0 = load_frag_k(kp + (long)kk * ld, ld);
    v16bf kb1 = load_frag_k(kp + (long)kk * ld + 32, ld);
    v16bf kb2 = load_frag_k(kp + (long)(kk + 16) * ld, ld);
    v16bf kb3 = load_frag_k(kp + (long)(kk + 16) * ld + 32, ld);
    v16bf vb[4];
#pragma unroll
    for (int j = 0; j < 4; ++j) vb[j] = load_frag_n(vp + (long)kk * ld + j * 16, ld);

    v8f s0 = z, s1 = z;
    s0 = wmma_bf16(qa0, kb0, s0);
    s0 = wmma_bf16(qa1, kb1, s0);
    s1 = wmma_bf16(qa0, kb2, s1);
    s1 = wmma_bf16(qa1, kb3, s1);
#pragma unroll
    for (int r = 0; r < 8; ++r) {
      float a0 = s0[r] * scale, a1 = s1[r] * scale;
      float v = fmaxf(a0, a1);
#pragma unroll
      for (int off = 1; off < 16; off <<= 1) v = fmaxf(v, __shfl_xor(v, off, 32));
      float mnew = fmaxf(mrow[r], v);
      float alpha = __expf(mrow[r] - mnew);
      mrow[r] = mnew;
      float p0 = __expf(a0 - mnew);
      float p1 = __expf(a1 - mnew);
      float ps = p0 + p1;
#pragma unroll
      for (int off = 1; off < 16; off <<= 1) ps += __shfl_xor(ps, off, 32);
      lrow[r] = lrow[r] * alpha + ps;
#pragma unroll
      for (int j = 0; j < 4; ++j) oacc[j][r] *= alpha;
      int row = r + ((lane & 16) ? 8 : 0);            // C-layout -> LDS relayout
      pbuf[wave][row * 32 + (lane & 15)]      = f2bf(p0);
      pbuf[wave][row * 32 + 16 + (lane & 15)] = f2bf(p1);
    }
    v16bf pa = load_frag_k(&pbuf[wave][0], 32);        // P as A-fragment
#pragma unroll
    for (int j = 0; j < 4; ++j) oacc[j] = wmma_bf16(pa, vb[j], oacc[j]);
  }
  int col = lane & 15, rb = (lane & 16) ? 8 : 0;
#pragma unroll
  for (int r = 0; r < 8; ++r) {
    float inv = 1.0f / lrow[r];
#pragma unroll
    for (int j = 0; j < 4; ++j) {
      long tok = (long)b * Tt + qt * 16 + rb + r;
      out[tok * Dm + h * HDh + j * 16 + col] = f2bf(oacc[j][r] * inv);
    }
  }
}

// ---------------------------------------------------------------- residual + LayerNorm
__global__ __launch_bounds__(256) void k_ln(float* __restrict__ x,
                                            const float* __restrict__ add,
                                            const float* __restrict__ g,
                                            const float* __restrict__ bb,
                                            bf16_t* __restrict__ xb) {
  long t = blockIdx.x;
  int tid = threadIdx.x;
  __shared__ float rsum[8], rsq[8];
  const float* xi = x + t * Dm;
  const float* ai = add + t * Dm;
  float loc[4];
  float s = 0.0f, sq = 0.0f;
#pragma unroll
  for (int i = 0; i < 4; ++i) {
    int d = tid + i * 256;
    float v = xi[d] + ai[d];
    loc[i] = v; s += v; sq += v * v;
  }
#pragma unroll
  for (int off = 1; off < 32; off <<= 1) {
    s += __shfl_xor(s, off, 32);
    sq += __shfl_xor(sq, off, 32);
  }
  int wave = tid >> 5, lane = tid & 31;
  if (lane == 0) { rsum[wave] = s; rsq[wave] = sq; }
  __syncthreads();
  float ts = 0.0f, tq = 0.0f;
#pragma unroll
  for (int w = 0; w < 8; ++w) { ts += rsum[w]; tq += rsq[w]; }
  float mu = ts * (1.0f / Dm);
  float var = tq * (1.0f / Dm) - mu * mu;
  float rstd = rsqrtf(var + 1e-5f);
  float* xo = x + t * Dm;
#pragma unroll
  for (int i = 0; i < 4; ++i) {
    int d = tid + i * 256;
    float y = (loc[i] - mu) * rstd * g[d] + bb[d];
    xo[d] = y;
    xb[t * Dm + d] = f2bf(y);
  }
}

// ---------------------------------------------------------------- router scores
__global__ __launch_bounds__(256) void k_router(const float* __restrict__ x,
                                                const float* __restrict__ rw,
                                                float* __restrict__ sc) {
  int tok = blockIdx.x * 8 + (threadIdx.x >> 5);
  int lane = threadIdx.x & 31;
  const float* xi = x + (long)tok * Dm;
  float s = 0.0f;
  for (int d = lane; d < Dm; d += 32) s += xi[d] * rw[d];
#pragma unroll
  for (int off = 1; off < 32; off <<= 1) s += __shfl_xor(s, off, 32);
  if (lane == 0) sc[tok] = s;
}

// ---------------------------------------------------------------- per-seq top-k (bitonic sort)
__global__ __launch_bounds__(1024) void k_topk(const float* __restrict__ sc,
                                               int* __restrict__ idx,
                                               float* __restrict__ gate) {
  __shared__ float s[Tm];
  __shared__ int   id[Tm];
  int b = blockIdx.x;
  for (int i = threadIdx.x; i < Tm; i += 1024) { s[i] = sc[(long)b * Tm + i]; id[i] = i; }
  __syncthreads();
  for (int ksz = 2; ksz <= Tm; ksz <<= 1) {
    for (int j = ksz >> 1; j > 0; j >>= 1) {
      for (int i = threadIdx.x; i < Tm; i += 1024) {
        int ixj = i ^ j;
        if (ixj > i) {
          float a = s[i], c = s[ixj];
          int ia = id[i], ic = id[ixj];
          bool aFirst = (a > c) || (a == c && ia < ic);   // descending rank
          bool descRegion = ((i & ksz) == 0);
          bool doSwap = descRegion ? (!aFirst) : aFirst;
          if (doSwap) { s[i] = c; s[ixj] = a; id[i] = ic; id[ixj] = ia; }
        }
      }
      __syncthreads();
    }
  }
  if (threadIdx.x < KSEL) {
    idx[b * KSEL + threadIdx.x] = id[threadIdx.x];
    gate[b * KSEL + threadIdx.x] = 1.0f / (1.0f + __expf(-s[threadIdx.x]));
  }
}

// ---------------------------------------------------------------- gather / scatter
__global__ __launch_bounds__(256) void k_gather(const float* __restrict__ x,
                                                const int* __restrict__ idx,
                                                float* __restrict__ xsel,
                                                bf16_t* __restrict__ xselb) {
  int j = blockIdx.x;
  int b = j >> 10;
  int t = idx[j];
  const float* src = x + ((long)b * Tm + t) * Dm;
  for (int d = threadIdx.x; d < Dm; d += 256) {
    float v = src[d];
    xsel[(long)j * Dm + d] = v;
    xselb[(long)j * Dm + d] = f2bf(v);
  }
}

__global__ __launch_bounds__(256) void k_scatter(float* __restrict__ x,
                                                 const float* __restrict__ proc,
                                                 const int* __restrict__ idx,
                                                 const float* __restrict__ gate) {
  int j = blockIdx.x;
  int b = j >> 10;
  int t = idx[j];
  float w = gate[j];
  float* dst = x + ((long)b * Tm + t) * Dm;
  const float* p = proc + (long)j * Dm;
  for (int d = threadIdx.x; d < Dm; d += 256) {
    float old = dst[d];
    dst[d] = old + (p[d] - old) * w;
  }
}

// ---------------------------------------------------------------- host orchestration
static void run_encoder(hipStream_t st, float* x, bf16_t* xb, int M, int Tt,
                        const bf16_t* wqkv, const float* bqkv,
                        const bf16_t* wo, const float* bo,
                        const bf16_t* w1, const float* b1,
                        const bf16_t* w2, const float* b2,
                        const float* g1, const float* be1,
                        const float* g2, const float* be2,
                        bf16_t* qkvb, bf16_t* ob16, bf16_t* h1b, float* tmpF) {
  dim3 blk(256);
  k_gemm<<<dim3(3 * Dm / 256, M / 64), blk, 0, st>>>(xb, wqkv, bqkv, nullptr, qkvb,
                                                     M, 3 * Dm, Dm, 0);
  int waves = Bm * Hh * (Tt / 16);
  k_attn<<<dim3(waves / 8), blk, 0, st>>>(qkvb, ob16, Tt);
  k_gemm<<<dim3(Dm / 256, M / 64), blk, 0, st>>>(ob16, wo, bo, tmpF, nullptr,
                                                 M, Dm, Dm, 0);
  k_ln<<<dim3(M), blk, 0, st>>>(x, tmpF, g1, be1, xb);
  k_gemm<<<dim3(DFFm / 256, M / 64), blk, 0, st>>>(xb, w1, b1, nullptr, h1b,
                                                   M, DFFm, Dm, 1);
  k_gemm<<<dim3(Dm / 256, M / 64), blk, 0, st>>>(h1b, w2, b2, tmpF, nullptr,
                                                 M, Dm, DFFm, 0);
  k_ln<<<dim3(M), blk, 0, st>>>(x, tmpF, g2, be2, xb);
}

extern "C" void kernel_launch(void* const* d_in, const int* in_sizes, int n_in,
                              void* d_out, int out_size, void* d_ws, size_t ws_size,
                              hipStream_t stream) {
  (void)in_sizes; (void)n_in; (void)out_size; (void)ws_size;
  const float* x_in = (const float*)d_in[0];
  const float* Wqkv = (const float*)d_in[1];
  const float* bqkv = (const float*)d_in[2];
  const float* Wo   = (const float*)d_in[3];
  const float* bo   = (const float*)d_in[4];
  const float* W1   = (const float*)d_in[5];
  const float* b1   = (const float*)d_in[6];
  const float* W2   = (const float*)d_in[7];
  const float* b2   = (const float*)d_in[8];
  const float* ln1g = (const float*)d_in[9];
  const float* ln1b = (const float*)d_in[10];
  const float* ln2g = (const float*)d_in[11];
  const float* ln2b = (const float*)d_in[12];
  const float* rw   = (const float*)d_in[13];

  float* x = (float*)d_out;  // running activations live in the output buffer

  // workspace carve-up
  char* wsb = (char*)d_ws;
  size_t o = 0;
  bf16_t* xb16   = (bf16_t*)(wsb + o); o += (size_t)Bm * Tm * Dm * 2;        // 16 MB
  bf16_t* qkvb   = (bf16_t*)(wsb + o); o += (size_t)Bm * Tm * 3 * Dm * 2;    // 48 MB
  bf16_t* ob16   = (bf16_t*)(wsb + o); o += (size_t)Bm * Tm * Dm * 2;        // 16 MB
  bf16_t* h1b    = (bf16_t*)(wsb + o); o += (size_t)Bm * Tm * DFFm * 2;      // 64 MB
  float*  tmpF   = (float*) (wsb + o); o += (size_t)Bm * Tm * Dm * 4;        // 32 MB
  float*  xsel   = (float*) (wsb + o); o += (size_t)Bm * KSEL * Dm * 4;      // 16 MB
  bf16_t* wbf    = (bf16_t*)(wsb + o); o += (size_t)(3 * Dm * Dm + Dm * Dm +
                                                     DFFm * Dm + Dm * DFFm) * 2; // 24 MB
  float*  scores = (float*) (wsb + o); o += (size_t)Bm * Tm * 4;
  int*    idxb   = (int*)   (wsb + o); o += (size_t)Bm * KSEL * 4;
  float*  gateb  = (float*) (wsb + o); o += (size_t)Bm * KSEL * 4;

  bf16_t* wq_b = wbf;
  bf16_t* wo_b = wbf + (size_t)3 * Dm * Dm;
  bf16_t* w1_b = wo_b + (size_t)Dm * Dm;
  bf16_t* w2_b = w1_b + (size_t)DFFm * Dm;

  hipMemcpyAsync(x, x_in, (size_t)Bm * Tm * Dm * sizeof(float),
                 hipMemcpyDeviceToDevice, stream);

  const size_t szQ = (size_t)3 * Dm * Dm;
  const size_t szO = (size_t)Dm * Dm;
  const size_t sz1 = (size_t)DFFm * Dm;
  const size_t sz2 = (size_t)Dm * DFFm;

  for (int i = 0; i < NLm; ++i) {
    k_cvt<<<dim3(2048), 256, 0, stream>>>(Wqkv + i * szQ, wq_b, (long)szQ);
    k_cvt<<<dim3(1024), 256, 0, stream>>>(Wo   + i * szO, wo_b, (long)szO);
    k_cvt<<<dim3(2048), 256, 0, stream>>>(W1   + i * sz1, w1_b, (long)sz1);
    k_cvt<<<dim3(2048), 256, 0, stream>>>(W2   + i * sz2, w2_b, (long)sz2);

    if (i & 1) {
      // ---- Mixture-of-Depths layer: route, gather, encode selected, scatter-gate
      k_router<<<dim3(Bm * Tm / 8), 256, 0, stream>>>(x, rw + (size_t)i * Dm, scores);
      k_topk<<<dim3(Bm), 1024, 0, stream>>>(scores, idxb, gateb);
      k_gather<<<dim3(Bm * KSEL), 256, 0, stream>>>(x, idxb, xsel, xb16);
      run_encoder(stream, xsel, xb16, Bm * KSEL, KSEL,
                  wq_b, bqkv + (size_t)i * 3 * Dm,
                  wo_b, bo + (size_t)i * Dm,
                  w1_b, b1 + (size_t)i * DFFm,
                  w2_b, b2 + (size_t)i * Dm,
                  ln1g + (size_t)i * Dm, ln1b + (size_t)i * Dm,
                  ln2g + (size_t)i * Dm, ln2b + (size_t)i * Dm,
                  qkvb, ob16, h1b, tmpF);
      k_scatter<<<dim3(Bm * KSEL), 256, 0, stream>>>(x, xsel, idxb, gateb);
    } else {
      // ---- full layer on all 8192 tokens
      k_cvt<<<dim3(2048), 256, 0, stream>>>(x, xb16, (long)Bm * Tm * Dm);
      run_encoder(stream, x, xb16, Bm * Tm, Tm,
                  wq_b, bqkv + (size_t)i * 3 * Dm,
                  wo_b, bo + (size_t)i * Dm,
                  w1_b, b1 + (size_t)i * DFFm,
                  w2_b, b2 + (size_t)i * Dm,
                  ln1g + (size_t)i * Dm, ln1b + (size_t)i * Dm,
                  ln2g + (size_t)i * Dm, ln2b + (size_t)i * Dm,
                  qkvb, ob16, h1b, tmpF);
    }
  }
}
